// Attention_22565758173369
// MI455X (gfx1250) — matrix operand order repaired and checked by reading, not compiled, not run on hardware
//
#include <hip/hip_runtime.h>
#include <hip/hip_bf16.h>

// ---------------------------------------------------------------------------
// Problem constants (match reference)
// ---------------------------------------------------------------------------
constexpr int BATCH    = 2;
constexpr int TSEQ     = 2048;
constexpr int HIDDEN   = 2048;
constexpr int NH       = 16;    // query heads
constexpr int NKV      = 4;     // kv heads
constexpr int HD       = 128;   // head dim
constexpr int NGROUPS  = NH / NKV;
constexpr float ROPE_THETA = 10000.0f;
constexpr int MT = BATCH * TSEQ;          // 4096 GEMM rows
constexpr float SM_SCALE = 0.08838834764831845f; // 1/sqrt(128)

typedef __attribute__((ext_vector_type(16))) _Float16 v16h;
typedef __attribute__((ext_vector_type(8)))  _Float16 v8h;
typedef __attribute__((ext_vector_type(8)))  float    v8f;

union H16 { v16h v; v8h h[2]; };

typedef __attribute__((address_space(3))) _Float16 lds_h;

// gfx1250 async copy: global -> LDS, 16 bytes per lane, tracked by ASYNCcnt.
__device__ __forceinline__ void async_copy_b128(_Float16* lds_dst, const _Float16* gsrc) {
    lds_h* ldsp = (lds_h*)lds_dst;                       // 32-bit LDS pointer
    asm volatile("global_load_async_to_lds_b128 %0, %1, off"
                 :: "v"(ldsp), "v"(gsrc) : "memory");
}
__device__ __forceinline__ void async_wait0() {
    asm volatile("s_wait_asynccnt 0x0" ::: "memory");
}

// ---------------------------------------------------------------------------
// WMMA GEMM:  C(f32, MxN) = A(MxK) * B(f32, KxN)
// Block tile 128x64, K-step 32, 256 threads = 8 waves (4M x 2N), each wave
// computes 32x32 via 4 WMMAs per K-step. Double-buffered LDS: global loads
// for step k+1 are register-staged while WMMAs consume buffer(k); one
// s_barrier per K-step.
// Fragment layouts (v_wmma_f32_16x16x32_f16, wave32):
//   A frag: lane m=lane%16, two 8-half runs at k = hlf*8 and hlf*8+16
//   B frag: lane n=lane%16, two 8-half runs at k = hlf*8 and hlf*8+16 (B staged ^T)
// ---------------------------------------------------------------------------
template <bool AF16>
__global__ __launch_bounds__(256)
void gemm_wmma(const void* __restrict__ Aptr, const float* __restrict__ Bw,
               float* __restrict__ C, int M, int N, int K)
{
    __shared__ __attribute__((aligned(16))) _Float16 As[2][128][40]; // [m][k]
    __shared__ __attribute__((aligned(16))) _Float16 Bs[2][64][40];  // [n][k] (^T)

    const int tid    = threadIdx.x;
    const int lane   = tid & 31;
    const int wid    = tid >> 5;       // 0..7
    const int lane16 = lane & 15;
    const int hlf    = lane >> 4;      // 0/1
    const int wm     = wid & 3;        // 32-row slab
    const int wn     = wid >> 2;       // 32-col slab

    const int row0 = blockIdx.y * 128;
    const int col0 = blockIdx.x * 64;

    v8f acc[2][2];
    #pragma unroll
    for (int i = 0; i < 2; ++i)
        #pragma unroll
        for (int j = 0; j < 2; ++j)
            #pragma unroll
            for (int r = 0; r < 8; ++r) acc[i][j][r] = 0.f;

    // staging assignments
    const int arow = tid >> 1;          // 0..127
    const int acol = (tid & 1) * 16;    // 0 or 16
    const int bk   = tid >> 3;          // 0..31
    const int bn   = (tid & 7) * 8;     // 0..56

    auto loadA = [&](int k0, v8h& r0, v8h& r1) {
        if (AF16) {
            const _Float16* p = (const _Float16*)Aptr + (size_t)(row0 + arow) * K + k0 + acol;
            r0 = *(const v8h*)p;
            r1 = *(const v8h*)(p + 8);
        } else {
            const float* p = (const float*)Aptr + (size_t)(row0 + arow) * K + k0 + acol;
            #pragma unroll
            for (int j = 0; j < 8; ++j) { r0[j] = (_Float16)p[j]; r1[j] = (_Float16)p[8 + j]; }
        }
    };
    auto loadB = [&](int k0, v8h& r) {
        const float* p = Bw + (size_t)(k0 + bk) * N + col0 + bn;
        #pragma unroll
        for (int j = 0; j < 8; ++j) r[j] = (_Float16)p[j];
    };
    auto storeA = [&](int buf, v8h r0, v8h r1) {
        *(v8h*)&As[buf][arow][acol]     = r0;
        *(v8h*)&As[buf][arow][acol + 8] = r1;
    };
    auto storeB = [&](int buf, v8h r) {
        #pragma unroll
        for (int j = 0; j < 8; ++j) Bs[buf][bn + j][bk] = r[j];
    };

    // prologue: fill buffer 0
    v8h pa0, pa1, pb;
    loadA(0, pa0, pa1);
    loadB(0, pb);
    storeA(0, pa0, pa1);
    storeB(0, pb);
    __syncthreads();

    for (int k0 = 0; k0 < K; k0 += 32) {
        const int  cur  = (k0 >> 5) & 1;
        const bool more = (k0 + 32) < K;

        // register-stage next tile while computing this one
        if (more) { loadA(k0 + 32, pa0, pa1); loadB(k0 + 32, pb); }

        H16 af[2], bf[2];
        #pragma unroll
        for (int i = 0; i < 2; ++i) {
            af[i].h[0] = *(const v8h*)&As[cur][wm*32 + i*16 + lane16][hlf*8];
            af[i].h[1] = *(const v8h*)&As[cur][wm*32 + i*16 + lane16][hlf*8 + 16];
            bf[i].h[0] = *(const v8h*)&Bs[cur][wn*32 + i*16 + lane16][hlf*8];
            bf[i].h[1] = *(const v8h*)&Bs[cur][wn*32 + i*16 + lane16][hlf*8 + 16];
        }
        #pragma unroll
        for (int i = 0; i < 2; ++i)
            #pragma unroll
            for (int j = 0; j < 2; ++j)
                acc[i][j] = __builtin_amdgcn_wmma_f32_16x16x32_f16(
                    false, af[i].v, false, bf[j].v, (short)0, acc[i][j], false, false);

        if (more) { storeA(cur ^ 1, pa0, pa1); storeB(cur ^ 1, pb); }
        __syncthreads();
    }

    // writeback: C layout VGPR r -> row r + hlf*8, col = lane16
    #pragma unroll
    for (int i = 0; i < 2; ++i)
        #pragma unroll
        for (int j = 0; j < 2; ++j)
            #pragma unroll
            for (int r = 0; r < 8; ++r) {
                const int row = row0 + wm*32 + i*16 + r + hlf*8;
                const int col = col0 + wn*32 + j*16 + lane16;
                C[(size_t)row * N + col] = acc[i][j][r];
            }
}

// ---------------------------------------------------------------------------
// RoPE + repack to head-major f16:
//   Qh[b][h][t][d], Kh[b][kv][t][d], Vh[b][kv][t][d].  One block per (b,t).
// ---------------------------------------------------------------------------
__global__ __launch_bounds__(256)
void rope_pack(const float* __restrict__ Qraw, const float* __restrict__ Kraw,
               const float* __restrict__ Vraw,
               _Float16* __restrict__ Qh, _Float16* __restrict__ Kh,
               _Float16* __restrict__ Vh)
{
    const int bt = blockIdx.x;
    const int b  = bt / TSEQ;
    const int t  = bt % TSEQ;
    const int tid = threadIdx.x;

    for (int p = tid; p < NH * (HD/2); p += 256) {
        const int h = p >> 6, i = p & 63;
        const float freq = __powf(ROPE_THETA, -(2.0f * (float)i) / (float)HD);
        float s, c; __sincosf((float)t * freq, &s, &c);
        const float* q = Qraw + (size_t)bt * HIDDEN + h * HD + 2 * i;
        const float x1 = q[0], x2 = q[1];
        _Float16* o = Qh + (((size_t)(b * NH + h)) * TSEQ + t) * HD + 2 * i;
        o[0] = (_Float16)(x1 * c - x2 * s);
        o[1] = (_Float16)(x1 * s + x2 * c);
    }
    for (int p = tid; p < NKV * (HD/2); p += 256) {
        const int h = p >> 6, i = p & 63;
        const float freq = __powf(ROPE_THETA, -(2.0f * (float)i) / (float)HD);
        float s, c; __sincosf((float)t * freq, &s, &c);
        const float* k = Kraw + (size_t)bt * (NKV * HD) + h * HD + 2 * i;
        const float x1 = k[0], x2 = k[1];
        _Float16* o = Kh + (((size_t)(b * NKV + h)) * TSEQ + t) * HD + 2 * i;
        o[0] = (_Float16)(x1 * c - x2 * s);
        o[1] = (_Float16)(x1 * s + x2 * c);
    }
    for (int e = tid; e < NKV * HD; e += 256) {
        const int kv = e >> 7, d = e & 127;
        Vh[(((size_t)(b * NKV + kv)) * TSEQ + t) * HD + d] =
            (_Float16)Vraw[(size_t)bt * (NKV * HD) + e];
    }
}

// ---------------------------------------------------------------------------
// Flash attention: 128 threads = 4 waves per block; each wave owns a 16-query
// tile of a 64-query span of one (b, head). The 4 waves cooperatively stage
// each 64-key chunk: K via gfx1250 async global->LDS DMA (ASYNCcnt), V via a
// register transpose so P*V B-fragments are contiguous ds_load_b128s. Chunk
// size == block query span, so every wave is partially unmasked in every
// chunk (no divergence; causality is pure masking).
// ---------------------------------------------------------------------------
__global__ __launch_bounds__(128)
void attn_flash(const _Float16* __restrict__ Qh, const _Float16* __restrict__ Kh,
                const _Float16* __restrict__ Vh, _Float16* __restrict__ Oh)
{
    __shared__ __attribute__((aligned(16))) _Float16 Ks[64][136];   // [key][dh]
    __shared__ __attribute__((aligned(16))) _Float16 Vt[HD][72];    // [dh][key] (^T)
    __shared__ __attribute__((aligned(16))) _Float16 Pt[4][16][72]; // per-wave [m][key]

    int bid = blockIdx.x;
    const int nqb = TSEQ / 64;            // 32 query-spans
    const int q64 = bid % nqb;  bid /= nqb;
    const int g   = bid % NGROUPS; bid /= NGROUPS;
    const int kv  = bid % NKV;
    const int b   = bid / NKV;
    const int h   = kv * NGROUPS + g;

    const int tid    = threadIdx.x;
    const int lane   = tid & 31;
    const int w      = tid >> 5;          // wave id 0..3
    const int lane16 = lane & 15;
    const int hlf    = lane >> 4;
    const int qbase  = q64 * 64 + w * 16; // this wave's query tile

    const _Float16* Qb = Qh + ((size_t)(b * NH  + h )) * TSEQ * HD;
    const _Float16* Kb = Kh + ((size_t)(b * NKV + kv)) * TSEQ * HD;
    const _Float16* Vb = Vh + ((size_t)(b * NKV + kv)) * TSEQ * HD;

    // Q fragments for the whole tile (4 dh-chunks of 32), kept in VGPRs.
    H16 aq[4];
    const _Float16* qrow = Qb + (size_t)(qbase + lane16) * HD;
    #pragma unroll
    for (int kc = 0; kc < 4; ++kc) {
        aq[kc].h[0] = *(const v8h*)(qrow + kc*32 + hlf*8);
        aq[kc].h[1] = *(const v8h*)(qrow + kc*32 + 16 + hlf*8);
    }

    v8f oacc[8];
    #pragma unroll
    for (int n = 0; n < 8; ++n)
        #pragma unroll
        for (int r = 0; r < 8; ++r) oacc[n][r] = 0.f;
    float rowmax[8], rowsum[8];
    #pragma unroll
    for (int r = 0; r < 8; ++r) { rowmax[r] = -1.0e30f; rowsum[r] = 0.f; }

    // staging assignments (128 threads, 64-key chunk)
    const int srow = tid >> 1;            // 0..63  (key row)
    const int sseg = (tid & 1) * 64;      // dh segment 0 or 64

    const int nchunks = q64 + 1;
    for (int ch = 0; ch < nchunks; ++ch) {
        const int kbase = ch * 64;

        // ---- K rows: async global->LDS DMA (no VGPR data path) ----
        {
            const _Float16* krow = Kb + (size_t)(kbase + srow) * HD + sseg;
            #pragma unroll
            for (int d8 = 0; d8 < 8; ++d8)
                async_copy_b128(&Ks[srow][sseg + d8 * 8], krow + d8 * 8);
        }
        // ---- V rows: register transpose into Vt[dh][key] ----
        {
            const _Float16* vrow = Vb + (size_t)(kbase + srow) * HD + sseg;
            #pragma unroll
            for (int d8 = 0; d8 < 8; ++d8) {
                v8h vv = *(const v8h*)(vrow + d8 * 8);
                #pragma unroll
                for (int j = 0; j < 8; ++j) Vt[sseg + d8 * 8 + j][srow] = vv[j];
            }
        }
        if (ch + 1 < nchunks)  // gfx1250 global_prefetch_b8 for next chunk
            __builtin_prefetch(Vb + (size_t)(kbase + 64 + srow) * HD + sseg, 0, 1);
        async_wait0();         // K DMA done (per wave) before barrier
        __syncthreads();

        // ---- S = Q K^T  (4 key-subtiles of 16), K frags from LDS ----
        v8f sacc[4];
        #pragma unroll
        for (int kt = 0; kt < 4; ++kt) {
            H16 bkf[4];
            #pragma unroll
            for (int kc = 0; kc < 4; ++kc) {
                bkf[kc].h[0] = *(const v8h*)&Ks[kt*16 + lane16][kc*32 + hlf*8];
                bkf[kc].h[1] = *(const v8h*)&Ks[kt*16 + lane16][kc*32 + 16 + hlf*8];
            }
            v8f s = {0.f,0.f,0.f,0.f,0.f,0.f,0.f,0.f};
            #pragma unroll
            for (int kc = 0; kc < 4; ++kc)
                s = __builtin_amdgcn_wmma_f32_16x16x32_f16(false, aq[kc].v, false, bkf[kc].v,
                                                           (short)0, s, false, false);
            const int kg = kbase + kt*16 + lane16;
            #pragma unroll
            for (int r = 0; r < 8; ++r) {
                const int mg = qbase + r + hlf*8;
                const float sv = s[r] * SM_SCALE;
                s[r] = (kg > mg) ? -1.0e30f : sv;
            }
            sacc[kt] = s;
        }

        // ---- online softmax stats + P to (per-wave) LDS ----
        #pragma unroll
        for (int r = 0; r < 8; ++r) {
            float tmax = fmaxf(fmaxf(sacc[0][r], sacc[1][r]),
                               fmaxf(sacc[2][r], sacc[3][r]));
            #pragma unroll
            for (int mk = 1; mk < 16; mk <<= 1)
                tmax = fmaxf(tmax, __shfl_xor(tmax, mk, 16));
            const float nm   = fmaxf(rowmax[r], tmax);
            const float corr = __expf(rowmax[r] - nm);
            rowmax[r] = nm;
            rowsum[r] *= corr;
            #pragma unroll
            for (int n = 0; n < 8; ++n) oacc[n][r] *= corr;

            float ps = 0.f;
            #pragma unroll
            for (int kt = 0; kt < 4; ++kt) {
                const float p = __expf(sacc[kt][r] - nm);
                ps += p;
                Pt[w][r + hlf*8][kt*16 + lane16] = (_Float16)p;
            }
            #pragma unroll
            for (int mk = 1; mk < 16; mk <<= 1)
                ps += __shfl_xor(ps, mk, 16);
            rowsum[r] += ps;
        }

        // ---- O += P * V  (2 key-chunks of 32, 8 dh-subtiles) ----
        #pragma unroll
        for (int kc = 0; kc < 2; ++kc) {
            H16 ap;
            ap.h[0] = *(const v8h*)&Pt[w][lane16][kc*32 + hlf*8];
            ap.h[1] = *(const v8h*)&Pt[w][lane16][kc*32 + 16 + hlf*8];
            #pragma unroll
            for (int n = 0; n < 8; ++n) {
                H16 bv;
                bv.h[0] = *(const v8h*)&Vt[n*16 + lane16][kc*32 + hlf*8];
                bv.h[1] = *(const v8h*)&Vt[n*16 + lane16][kc*32 + 16 + hlf*8];
                oacc[n] = __builtin_amdgcn_wmma_f32_16x16x32_f16(false, ap.v, false, bv.v,
                                                                 (short)0, oacc[n], false, false);
            }
        }
        __syncthreads();   // protect Ks/Vt before next chunk's staging
    }

    // ---- normalize + write Oh[b][t][h][d] ----
    #pragma unroll
    for (int r = 0; r < 8; ++r) {
        const float inv = 1.0f / rowsum[r];
        const int t = qbase + r + hlf*8;
        _Float16* orow = Oh + (((size_t)b * TSEQ + t) * NH + h) * HD;
        #pragma unroll
        for (int n = 0; n < 8; ++n)
            orow[n*16 + lane16] = (_Float16)(oacc[n][r] * inv);
    }
}

// ---------------------------------------------------------------------------
// Host-side launch
// ---------------------------------------------------------------------------
extern "C" void kernel_launch(void* const* d_in, const int* in_sizes, int n_in,
                              void* d_out, int out_size, void* d_ws, size_t ws_size,
                              hipStream_t stream)
{
    (void)in_sizes; (void)n_in; (void)out_size; (void)ws_size;
    const float* x  = (const float*)d_in[0];
    const float* wq = (const float*)d_in[1];
    const float* wk = (const float*)d_in[2];
    const float* wv = (const float*)d_in[3];
    const float* wo = (const float*)d_in[4];
    float* out = (float*)d_out;

    char* ws = (char*)d_ws;
    size_t off = 0;
    auto carve = [&](size_t bytes) -> void* {
        void* p = ws + off;
        off += (bytes + 255) & ~(size_t)255;
        return p;
    };

    float*     Qraw = (float*)    carve((size_t)MT * HIDDEN   * sizeof(float));
    float*     Kraw = (float*)    carve((size_t)MT * NKV * HD * sizeof(float));
    float*     Vraw = (float*)    carve((size_t)MT * NKV * HD * sizeof(float));
    _Float16*  Qh   = (_Float16*) carve((size_t)MT * HIDDEN   * sizeof(_Float16));
    _Float16*  Kh   = (_Float16*) carve((size_t)MT * NKV * HD * sizeof(_Float16));
    _Float16*  Vh   = (_Float16*) carve((size_t)MT * NKV * HD * sizeof(_Float16));
    _Float16*  Oh   = (_Float16*) carve((size_t)MT * HIDDEN   * sizeof(_Float16));

    // QKV projections
    gemm_wmma<false><<<dim3(HIDDEN/64,  MT/128), 256, 0, stream>>>(x, wq, Qraw, MT, HIDDEN, HIDDEN);
    gemm_wmma<false><<<dim3((NKV*HD)/64, MT/128), 256, 0, stream>>>(x, wk, Kraw, MT, NKV*HD, HIDDEN);
    gemm_wmma<false><<<dim3((NKV*HD)/64, MT/128), 256, 0, stream>>>(x, wv, Vraw, MT, NKV*HD, HIDDEN);

    // RoPE + pack to f16 head-major
    rope_pack<<<BATCH * TSEQ, 256, 0, stream>>>(Qraw, Kraw, Vraw, Qh, Kh, Vh);

    // Flash attention: 4 waves per block, 64 queries per block
    attn_flash<<<BATCH * NKV * NGROUPS * (TSEQ/64), 128, 0, stream>>>(Qh, Kh, Vh, Oh);

    // Output projection
    gemm_wmma<true><<<dim3(HIDDEN/64, MT/128), 256, 0, stream>>>(Oh, wo, out, MT, HIDDEN, HIDDEN);
}
